// ClassicalSelfAttention_65481071407449
// MI455X (gfx1250) — matrix operand order repaired and checked by reading, not compiled, not run on hardware
//
#include <hip/hip_runtime.h>

// ---------------------------------------------------------------------------
// Classical self-attention (B=4, S=2048, E=1024, fp32 in/out) for gfx1250.
// Matmuls via V_WMMA_F32_16X16X32_BF16 (fp32 accumulate, bf16 operands).
// Big GEMMs use GLOBAL_LOAD_ASYNC_TO_LDS_B128 double-buffered tiles (ASYNCcnt)
// with ds_load_b128 fragment reads feeding the WMMA pipes.
// ---------------------------------------------------------------------------

#define BATCH   4
#define SEQ     2048
#define EMBED   1024
#define MROWS   (BATCH * SEQ)          // 8192

typedef __attribute__((ext_vector_type(16))) __bf16 v16bf;
typedef __attribute__((ext_vector_type(8)))  __bf16 v8bf;
typedef __attribute__((ext_vector_type(8)))  float  v8f;

// D = A(16x32 bf16) * B(32x16 bf16) + C(16x16 f32)
__device__ inline v8f wmma_bf16(v16bf a, v16bf b, v8f c) {
  return __builtin_amdgcn_wmma_f32_16x16x32_bf16(
      /*neg_a=*/false, a, /*neg_b=*/false, b,
      /*c_mod=*/(short)0, c, /*reuse_a=*/false, /*reuse_b=*/false);
}

__device__ inline v16bf cat16(v8bf lo, v8bf hi) {
  return __builtin_shufflevector(lo, hi, 0, 1, 2, 3, 4, 5, 6, 7,
                                 8, 9, 10, 11, 12, 13, 14, 15);
}

__device__ inline v16bf cvt16(v8f lo, v8f hi) {
  v16bf r;
#pragma unroll
  for (int i = 0; i < 8; ++i) {
    r[i]     = (__bf16)lo[i];
    r[i + 8] = (__bf16)hi[i];
  }
  return r;
}

__device__ inline void wait_async0() {
#if __has_builtin(__builtin_amdgcn_s_wait_asynccnt)
  __builtin_amdgcn_s_wait_asynccnt(0);
#else
  asm volatile("s_wait_asynccnt 0" ::: "memory");
#endif
}

// One lane moves 16B global -> LDS, tracked by ASYNCcnt.
__device__ inline void async_copy_b128(unsigned lds_byte_addr, const void* gptr) {
  asm volatile("global_load_async_to_lds_b128 %0, %1, off"
               :: "v"(lds_byte_addr), "v"(gptr) : "memory");
}

// ---- Global-direct fragment loaders (fp32 source, used by proj kernel) ----
// A fragment (16x32 MxK), ISA 7.12.2 16-bit A layout.
__device__ inline v16bf load_a_f32(const float* __restrict__ base, int ld, int k0) {
  const int lane = threadIdx.x & 31;
  const float* p = base + (size_t)(lane & 15) * ld + k0 + ((lane >> 4) << 3);
  v8f lo = *(const v8f*)p;
  v8f hi = *(const v8f*)(p + 16);
  return cvt16(lo, hi);
}
// B fragment (32x16 KxN), B[k][n] = rows_n[k].
__device__ inline v16bf load_b_f32(const float* __restrict__ rows, int ld, int k0) {
  const int lane = threadIdx.x & 31;
  const float* p = rows + (size_t)(lane & 15) * ld + k0 + ((lane >> 4) << 3);
  v8f lo = *(const v8f*)p;
  v8f hi = *(const v8f*)(p + 16);
  return cvt16(lo, hi);
}

// ---- LDS fragment loaders (bf16 tiles, padded rows of 40 elements) ----
#define LROW 40  // 32 K-elements + 8 pad => 80B stride, conflict-free b128 reads
typedef __bf16 ldsrow_t[LROW];

__device__ inline v16bf lds_frag_a(const ldsrow_t* rows, int r0) {
  const int lane = threadIdx.x & 31;
  const __bf16* p = rows[r0 + (lane & 15)] + ((lane >> 4) << 3);
  v8bf lo = *(const v8bf*)p;
  v8bf hi = *(const v8bf*)(p + 16);
  return cat16(lo, hi);
}
__device__ inline v16bf lds_frag_b(const ldsrow_t* rows, int n0) {
  const int lane = threadIdx.x & 31;
  const __bf16* p = rows[n0 + (lane & 15)] + ((lane >> 4) << 3);
  v8bf lo = *(const v8bf*)p;
  v8bf hi = *(const v8bf*)(p + 16);
  return cat16(lo, hi);
}

// ---------------------------------------------------------------------------
// Kernel 1: projection GEMM  out = x @ W^T   (M=8192, N=1024, K=1024), fp32 in.
// mode 0: bf16 row-major, scaled (Q/sqrt(E));  1: bf16 row-major (K);
// mode 2: bf16 transposed per batch Vt[b][f][s].
// ---------------------------------------------------------------------------
__global__ __launch_bounds__(256) void proj_kernel(
    const float* __restrict__ X, const float* __restrict__ W,
    __bf16* __restrict__ out, int mode, float scale) {
  const int wave = threadIdx.x >> 5;
  const int m0 = blockIdx.x * 128 + wave * 16;
  const int n0 = blockIdx.y * 64;

  v8f acc[4] = {};
  const float* Xrow = X + (size_t)m0 * EMBED;

#pragma unroll 2
  for (int k0 = 0; k0 < EMBED; k0 += 32) {
    v16bf a = load_a_f32(Xrow, EMBED, k0);
#pragma unroll
    for (int t = 0; t < 4; ++t) {
      v16bf b = load_b_f32(W + (size_t)(n0 + t * 16) * EMBED, EMBED, k0);
      acc[t] = wmma_bf16(a, b, acc[t]);
    }
  }

  const int lane  = threadIdx.x & 31;
  const int ncol  = lane & 15;
  const int mbase = (lane >> 4) << 3;
#pragma unroll
  for (int t = 0; t < 4; ++t) {
    const int n = n0 + t * 16 + ncol;
#pragma unroll
    for (int v = 0; v < 8; ++v) {
      const int m = m0 + mbase + v;
      const float val = acc[t][v] * scale;
      if (mode == 2) {
        const int bi = m >> 11;
        const int s  = m & (SEQ - 1);
        out[((size_t)(bi * EMBED + n) << 11) + s] = (__bf16)val;
      } else {
        out[(size_t)m * EMBED + n] = (__bf16)val;
      }
    }
  }
}

// ---------------------------------------------------------------------------
// Kernel 2: NT GEMM, bf16 x bf16 -> f32:  C[m][n] = sum_k A[m][k]*B[n][k]
// Used for scores = Qs@K^T (kdim=1024) and out = P@Vt (kdim=2048).
// Block 256 thr (8 waves): tile 256M x 64N; wave tile 32M x 64N (8 wmma/step).
// K pipelined in 32-element steps, double-buffered async-to-LDS staging.
// ---------------------------------------------------------------------------
__global__ __launch_bounds__(256) void gemm_nt_bf16_kernel(
    const __bf16* __restrict__ A, const __bf16* __restrict__ Bm,
    float* __restrict__ C, int kdim, int ldc,
    size_t strideA, size_t strideB, size_t strideC) {
  __shared__ __bf16 As[2][256][LROW];  // 40 KB
  __shared__ __bf16 Bs[2][64][LROW];   // 10 KB

  const int b    = blockIdx.z;
  const int wave = threadIdx.x >> 5;
  const int tid  = threadIdx.x;
  const int m0 = blockIdx.x * 256;
  const int n0 = blockIdx.y * 64;

  const __bf16* Abase = A  + b * strideA + (size_t)m0 * kdim;
  const __bf16* Bbase = Bm + b * strideB + (size_t)n0 * kdim;

  // Issue async fills for one K-step tile: A 256x32 (4 chunks/thr), B 64x32 (1).
  auto issue_tiles = [&](int bsel, int k0) {
#pragma unroll
    for (int j = 0; j < 4; ++j) {
      const int ci = tid + j * 256;        // 0..1023
      const int ra = ci >> 2, c = ci & 3;  // row, 16B-chunk
      const __bf16* g = Abase + (size_t)ra * kdim + k0 + c * 8;
      async_copy_b128((unsigned)(uintptr_t)&As[bsel][ra][c * 8], g);
    }
    const int rb = tid >> 2, cb = tid & 3;
    const __bf16* g = Bbase + (size_t)rb * kdim + k0 + cb * 8;
    async_copy_b128((unsigned)(uintptr_t)&Bs[bsel][rb][cb * 8], g);
  };

  v8f acc[2][4] = {};
  issue_tiles(0, 0);

  int buf = 0;
  for (int k0 = 0; k0 < kdim; k0 += 32) {
    wait_async0();        // my async fills of buf are done
    __syncthreads();      // everyone's fills done; buf^1 free of readers
    if (k0 + 32 < kdim) issue_tiles(buf ^ 1, k0 + 32);

    const v16bf a0 = lds_frag_a(As[buf], wave * 32);
    const v16bf a1 = lds_frag_a(As[buf], wave * 32 + 16);
#pragma unroll
    for (int t = 0; t < 4; ++t) {
      const v16bf bb = lds_frag_b(Bs[buf], t * 16);
      acc[0][t] = wmma_bf16(a0, bb, acc[0][t]);
      acc[1][t] = wmma_bf16(a1, bb, acc[1][t]);
    }
    buf ^= 1;
  }

  float* Cb = C + b * strideC;
  const int lane  = threadIdx.x & 31;
  const int ncol  = lane & 15;
  const int mbase = (lane >> 4) << 3;
#pragma unroll
  for (int f = 0; f < 2; ++f) {
#pragma unroll
    for (int t = 0; t < 4; ++t) {
      const int n = n0 + t * 16 + ncol;
#pragma unroll
      for (int v = 0; v < 8; ++v) {
        const int m = m0 + wave * 32 + f * 16 + mbase + v;
        Cb[(size_t)m * ldc + n] = acc[f][t][v];
      }
    }
  }
}

// ---------------------------------------------------------------------------
// Kernel 3: row softmax over f32 scores -> bf16 P. One block per row.
// ---------------------------------------------------------------------------
__global__ __launch_bounds__(256) void softmax_kernel(
    const float* __restrict__ Sc, __bf16* __restrict__ P) {
  __shared__ float row[SEQ];
  __shared__ float sbuf[256];
  const size_t r = blockIdx.x;
  const float* src = Sc + r * SEQ;
  __bf16* dst = P + r * SEQ;
  const int tid = threadIdx.x;

  float lmax = -1e30f;
#pragma unroll
  for (int i = tid; i < SEQ; i += 256) {
    const float v = src[i];
    row[i] = v;
    lmax = fmaxf(lmax, v);
  }
  sbuf[tid] = lmax;
  __syncthreads();
  for (int s = 128; s > 0; s >>= 1) {
    if (tid < s) sbuf[tid] = fmaxf(sbuf[tid], sbuf[tid + s]);
    __syncthreads();
  }
  const float rmax = sbuf[0];
  __syncthreads();

  float lsum = 0.0f;
#pragma unroll
  for (int i = tid; i < SEQ; i += 256) {
    const float e = __expf(row[i] - rmax);
    row[i] = e;
    lsum += e;
  }
  sbuf[tid] = lsum;
  __syncthreads();
  for (int s = 128; s > 0; s >>= 1) {
    if (tid < s) sbuf[tid] += sbuf[tid + s];
    __syncthreads();
  }
  const float inv = 1.0f / sbuf[0];
#pragma unroll
  for (int i = tid; i < SEQ; i += 256) {
    dst[i] = (__bf16)(row[i] * inv);
  }
}

// ---------------------------------------------------------------------------
// Workspace layout (bytes):
//   Qs  bf16 [8192][1024]          16 MB   (pre-scaled by 1/sqrt(E))
//   Kb  bf16 [8192][1024]          16 MB
//   Vt  bf16 [4][1024][2048]       16 MB   (V transposed per batch)
//   Sc  f32  [4][2048][2048]       64 MB   (raw scores)
//   P   bf16 [4][2048][2048]       32 MB   (softmax probs)
// ---------------------------------------------------------------------------
extern "C" void kernel_launch(void* const* d_in, const int* in_sizes, int n_in,
                              void* d_out, int out_size, void* d_ws, size_t ws_size,
                              hipStream_t stream) {
  (void)in_sizes; (void)n_in; (void)out_size; (void)ws_size;
  const float* x  = (const float*)d_in[0];
  const float* Wq = (const float*)d_in[1];
  const float* Wk = (const float*)d_in[2];
  const float* Wv = (const float*)d_in[3];

  char* w = (char*)d_ws;
  const size_t SZ_PROJ = (size_t)MROWS * EMBED * 2;      // 16 MB
  const size_t SZ_SC   = (size_t)BATCH * SEQ * SEQ * 4;  // 64 MB
  __bf16* Qs  = (__bf16*)(w);
  __bf16* Kb  = (__bf16*)(w + SZ_PROJ);
  __bf16* Vt  = (__bf16*)(w + 2 * SZ_PROJ);
  float*  Sc  = (float*) (w + 3 * SZ_PROJ);
  __bf16* Pp  = (__bf16*)(w + 3 * SZ_PROJ + SZ_SC);

  const dim3 blk(256);
  const float inv_sqrt_d = 0.03125f;  // 1/sqrt(1024)

  // Q/K/V projections (fp32 -> bf16 WMMA GEMM)
  proj_kernel<<<dim3(MROWS / 128, EMBED / 64), blk, 0, stream>>>(x, Wq, Qs, 0, inv_sqrt_d);
  proj_kernel<<<dim3(MROWS / 128, EMBED / 64), blk, 0, stream>>>(x, Wk, Kb, 1, 1.0f);
  proj_kernel<<<dim3(MROWS / 128, EMBED / 64), blk, 0, stream>>>(x, Wv, Vt, 2, 1.0f);

  // scores = Qs @ K^T per batch  (M=2048, N=2048, K=1024)
  gemm_nt_bf16_kernel<<<dim3(SEQ / 256, SEQ / 64, BATCH), blk, 0, stream>>>(
      Qs, Kb, Sc, EMBED, SEQ,
      (size_t)SEQ * EMBED, (size_t)SEQ * EMBED, (size_t)SEQ * SEQ);

  // softmax rows
  softmax_kernel<<<dim3(MROWS), blk, 0, stream>>>(Sc, Pp);

  // out = P @ Vt per batch  (M=2048, N=1024, K=2048)
  gemm_nt_bf16_kernel<<<dim3(SEQ / 256, EMBED / 64, BATCH), blk, 0, stream>>>(
      Pp, Vt, (float*)d_out, SEQ, EMBED,
      (size_t)SEQ * SEQ, (size_t)EMBED * SEQ, (size_t)SEQ * EMBED);
}
